// BondCenteredTensorMomentDescriptor_61933428408904
// MI455X (gfx1250) — matrix operand order repaired and checked by reading, not compiled, not run on hardware
//
#include <hip/hip_runtime.h>
#include <hip/hip_bf16.h>
#include <hip/hip_fp16.h>

typedef __attribute__((ext_vector_type(16))) _Float16 v16h;
typedef __attribute__((ext_vector_type(8)))  float    v8f;

#define FDIM 128
#define CUTOFF_R 5.0f
#define NPATH 23
#define SA_STR 136            // 128 halves + 8 pad (4-bank rotation per row)
#define SC_STR 132            // 128 floats + 4 pad
#define K1_LDS (240*SA_STR*2 + 240*SC_STR*4)   // 65,280 + 126,720 = 192,000 B

// ---- constant tables -------------------------------------------------------
__constant__ int PL1[NPATH] = {0,0,0,0,0, 1,1,1,1,1,1,1,1, 2,2,2,2,2,2,2,2,2,2};
__constant__ int PL2[NPATH] = {0,1,2,3,4, 0,1,1,2,2,3,3,4, 0,1,1,2,2,2,3,3,4,4};
__constant__ int PL3[NPATH] = {0,1,2,3,4, 1,0,2,1,3,2,4,3, 2,1,3,0,2,4,1,3,2,4};

// padded row bases: l=0 rows[0,16) l=1 [16,48) l=2 [48,96) l=3 [96,160) l=4 [160,240)
__constant__ int P_L[5] = {0,16,48,96,160};
__constant__ int ROWBASE[25] = {0, 16,17,18, 48,49,50,51,52,
                                96,97,98,99,100,101,102,
                                160,161,162,163,164,165,166,167,168};
__constant__ int ROWMUL[25]  = {1, 3,3,3, 5,5,5,5,5, 7,7,7,7,7,7,7, 9,9,9,9,9,9,9,9,9};

struct Ent { float val; int packed; };

// ---- setup: real Clebsch-Gordan tables (mirrors the Python reference) ------
__device__ double dfact(int n) {
    const double f[13] = {1.,1.,2.,6.,24.,120.,720.,5040.,40320.,362880.,
                          3628800.,39916800.,479001600.};
    return f[n];
}

__device__ double cgc_dev(int l1,int l2,int l3,int m1,int m2,int m3) {
    if (m1 + m2 != m3 || l3 < abs(l1-l2) || l3 > l1+l2) return 0.0;
    if (abs(m3) > l3 || abs(m1) > l1 || abs(m2) > l2) return 0.0;
    double pre = sqrt((2.0*l3+1.0)*dfact(l3+l1-l2)*dfact(l3-l1+l2)*dfact(l1+l2-l3)
                      / dfact(l1+l2+l3+1));
    pre *= sqrt(dfact(l3+m3)*dfact(l3-m3)*dfact(l1-m1)*dfact(l1+m1)
                *dfact(l2-m2)*dfact(l2+m2));
    double s = 0.0;
    for (int k = 0; k <= l1+l2-l3; ++k) {
        int a0=k, a1=l1+l2-l3-k, a2=l1-m1-k, a3=l2+m2-k, a4=l3-l2+m1+k, a5=l3-l1-m2+k;
        if (a0<0||a1<0||a2<0||a3<0||a4<0||a5<0) continue;
        double d = dfact(a0)*dfact(a1)*dfact(a2)*dfact(a3)*dfact(a4)*dfact(a5);
        s += ((k & 1) ? -1.0 : 1.0) / d;
    }
    return pre * s;
}

__device__ void umat_dev(int l, int i, int j, double& re, double& im) {
    re = 0.0; im = 0.0;
    int mr = i - l, mc = j - l;
    const double is2 = 0.7071067811865476;
    if (mr == 0) { if (mc == 0) re = 1.0; return; }
    if (mr > 0) {
        double sgn = (mr & 1) ? -1.0 : 1.0;
        if (mc == mr)       re = sgn * is2;
        else if (mc == -mr) im = sgn * is2;
    } else {
        int m = -mr;
        if (mc == m)        re = is2;
        else if (mc == -m)  im = -is2;
    }
}

__global__ void k0_cg(float* dense) {
    int p = blockIdx.x;
    int l1 = PL1[p], l2 = PL2[p], l3 = PL3[p];
    int n1 = 2*l1+1, n2 = 2*l2+1, n3 = 2*l3+1;
    int sz = n1*n2*n3;
    int off = 0;
    for (int q = 0; q < p; ++q) off += (2*PL1[q]+1)*(2*PL2[q]+1)*(2*PL3[q]+1);
    int idx = threadIdx.x;
    if (idx >= sz) return;
    int C  = idx % n3;
    int AB = idx / n3;
    int B  = AB % n2;
    int A  = AB / n2;
    double re = 0.0, im = 0.0;
    for (int a = 0; a < n1; ++a) {
        for (int b = 0; b < n2; ++b) {
            int m3 = (a - l1) + (b - l2);
            if (m3 < -l3 || m3 > l3) continue;
            double cg = cgc_dev(l1, l2, l3, a - l1, b - l2, m3);
            if (cg == 0.0) continue;
            int c = l3 + m3;
            double r1,i1,r2,i2,r3,i3;
            umat_dev(l1, a, A, r1, i1);
            umat_dev(l2, b, B, r2, i2);
            umat_dev(l3, c, C, r3, i3);
            i3 = -i3; // conj(U3)
            double tr = r1*r2 - i1*i2, ti = r1*i2 + i1*r2;
            double ur = tr*r3 - ti*i3, ui = tr*i3 + ti*r3;
            re += cg * ur; im += cg * ui;
        }
    }
    (void)im;
    dense[off + idx] = (float)re;
}

__global__ void k0_compact(const float* dense, Ent* entries, int* nnz) {
    if (threadIdx.x != 0 || blockIdx.x != 0) return;
    int cnt = 0, off = 0;
    for (int p = 0; p < NPATH; ++p) {
        int l1 = PL1[p], l2 = PL2[p], l3 = PL3[p];
        int n1 = 2*l1+1, n2 = 2*l2+1, n3 = 2*l3+1;
        for (int A = 0; A < n1; ++A)
            for (int B = 0; B < n2; ++B)
                for (int C = 0; C < n3; ++C) {
                    float v = dense[off + (A*n2 + B)*n3 + C];
                    if (fabsf(v) > 1e-7f) {
                        int ag = l1*l1 + A, bg = l2*l2 + B, cg = l3*l3 + C;
                        entries[cnt].val = v;
                        entries[cnt].packed = ag | (bg << 8) | (cg << 16) | (p << 24);
                        ++cnt;
                    }
                }
        off += n1*n2*n3;
    }
    *nnz = cnt;
}

// ---- WMMA fragment helpers (ISA 7.12.2 layouts, wave32) --------------------
// 16-bit A 16x32: lanes 0-15 = rows, vgpr d halves 2d,2d+1 -> K pairs; hi lanes +8
__device__ inline v16h loadAfrag(const _Float16* sA, int rowBase, int k0, int lane) {
    int m = rowBase + (lane & 15);
    int koff = (lane & 16) ? 8 : 0;
    const _Float16* rp = sA + m * SA_STR + k0 + koff;
    union { v16h v; unsigned u[8]; } a;
#pragma unroll
    for (int d = 0; d < 8; ++d) {
        int kk = 2*d + ((d >= 4) ? 8 : 0);
        a.u[d] = *reinterpret_cast<const unsigned*>(rp + kk);
    }
    return a.v;
}

// 16-bit B 32x16: lane%16 = column n, half h -> K=h (h<8) / h+8 (h>=8), +8 for hi lanes
__device__ inline v16h loadBfrag(const float* W, int k0, int n0, int lane) {
    int n = n0 + (lane & 15);
    int kbase = k0 + ((lane & 16) ? 8 : 0);
    v16h b;
#pragma unroll
    for (int h = 0; h < 16; ++h)
        b[h] = (_Float16)W[(kbase + h + ((h >= 8) ? 8 : 0)) * FDIM + n];
    return b;
}

// ---- kernel 1: gather + edense(W1) + normalize + mish + edense(W2)+res ----
__global__ __launch_bounds__(256) void k1_edense(
        const float* __restrict__ Adesc, const int* __restrict__ nbr,
        const float* __restrict__ W1, const float* __restrict__ b1,
        const float* __restrict__ W2, const float* __restrict__ b2,
        float* __restrict__ yOut) {
    extern __shared__ char smem[];
    _Float16* sA = (_Float16*)smem;                       // 240 x SA_STR halves
    float*    sC = (float*)(smem + 240 * SA_STR * 2);     // 240 x SC_STR floats

    const int tid  = threadIdx.x;
    const int lane = tid & 31;
    const int wv   = tid >> 5;
    const int n0   = wv * 16;
    const int eBase = blockIdx.x * 8;

    // ---- stage A: gather y0 = A[i]+A[j] -> f16 LDS; zero padding rows ----
    for (int t = tid; t < 8*25*FDIM; t += 256) {
        int e  = t / 3200;
        int rm = t - e * 3200;
        int m  = rm >> 7;
        int f  = rm & 127;
        int ia = nbr[(eBase + e) * 2 + 0];
        int ja = nbr[(eBase + e) * 2 + 1];
        float v = Adesc[(size_t)ia * 3200 + m * FDIM + f]
                + Adesc[(size_t)ja * 3200 + m * FDIM + f];
        int row = ROWBASE[m] + e * ROWMUL[m];
        sA[row * SA_STR + f] = (_Float16)v;
    }
    for (int t = tid; t < 40*FDIM; t += 256) {   // 8 pad rows per l
        int pr = t >> 7, f = t & 127;
        int l = pr >> 3, q = pr & 7;
        int row = P_L[l] + 8 * (2*l + 1) + q;
        sA[row * SA_STR + f] = (_Float16)0.f;
    }
    __syncthreads();

    // ---- stage B: GEMM1 per l, C-init = b1 for l==0, result -> sC (f32) ----
#pragma unroll
    for (int l = 0; l < 5; ++l) {
        const float* Wl = W1 + l * FDIM * FDIM;
        v16h bf[4];
#pragma unroll
        for (int ks = 0; ks < 4; ++ks) bf[ks] = loadBfrag(Wl, ks * 32, n0, lane);
        float cinit = (l == 0) ? b1[n0 + (lane & 15)] : 0.f;
        for (int lt = 0; lt <= l; ++lt) {
            int rowBase = P_L[l] + lt * 16;
            v8f c;
#pragma unroll
            for (int r = 0; r < 8; ++r) c[r] = cinit;
#pragma unroll
            for (int ks = 0; ks < 4; ++ks) {
                v16h a = loadAfrag(sA, rowBase, ks * 32, lane);
                c = __builtin_amdgcn_wmma_f32_16x16x32_f16(
                        false, a, false, bf[ks], (short)0, c, false, false);
            }
            int n  = n0 + (lane & 15);
            int mo = (lane & 16) ? 8 : 0;
#pragma unroll
            for (int r = 0; r < 8; ++r)
                sC[(rowBase + r + mo) * SC_STR + n] = c[r];
        }
    }
    __syncthreads();

    // ---- stage C: normalize (over 25 m) + gated mish -> f16 back to sA ----
#pragma unroll
    for (int q = 0; q < 4; ++q) {
        int idx = tid + q * 256;        // 8 edges * 128 feats = 1024 pairs
        int e = idx >> 7, g = idx & 127;
        float vals[25]; float n2 = 0.f;
#pragma unroll
        for (int m = 0; m < 25; ++m) {
            float v = sC[(ROWBASE[m] + e * ROWMUL[m]) * SC_STR + g];
            vals[m] = v; n2 += v * v;
        }
        float inv = rsqrtf(n2 + 1e-6f);
        float s  = vals[0] * inv;
        float sp = (s > 20.f) ? s : log1pf(__expf(s));
        float th = tanhf(sp);
        float sg = 1.f / (1.f + __expf(-s));
        float dm = th + s * (1.f - th * th) * sg;
        sA[(ROWBASE[0] + e) * SA_STR + g] = (_Float16)(s * th);
#pragma unroll
        for (int m = 1; m < 25; ++m)
            sA[(ROWBASE[m] + e * ROWMUL[m]) * SA_STR + g] =
                (_Float16)(vals[m] * inv * dm);
    }
    __syncthreads();

    // ---- stage D: GEMM2, C-init = residual(+b2 for l==0), store y2 -> global
#pragma unroll
    for (int l = 0; l < 5; ++l) {
        const float* Wl = W2 + l * FDIM * FDIM;
        const int TW = 2 * l + 1;
        v16h bf[4];
#pragma unroll
        for (int ks = 0; ks < 4; ++ks) bf[ks] = loadBfrag(Wl, ks * 32, n0, lane);
        int n  = n0 + (lane & 15);
        int mo = (lane & 16) ? 8 : 0;
        float bb = (l == 0) ? b2[n] : 0.f;
        for (int lt = 0; lt <= l; ++lt) {
            int rowBase = P_L[l] + lt * 16;
            v8f c;
#pragma unroll
            for (int r = 0; r < 8; ++r)
                c[r] = (float)sA[(rowBase + r + mo) * SA_STR + n] + bb;
#pragma unroll
            for (int ks = 0; ks < 4; ++ks) {
                v16h a = loadAfrag(sA, rowBase, ks * 32, lane);
                c = __builtin_amdgcn_wmma_f32_16x16x32_f16(
                        false, a, false, bf[ks], (short)0, c, false, false);
            }
#pragma unroll
            for (int r = 0; r < 8; ++r) {
                int ro = lt * 16 + r + mo;           // row within this l block
                int e  = ro / TW;                    // TW is compile-time const
                int ml = ro - e * TW;
                if (e < 8)
                    yOut[(size_t)(eBase + e) * 3200 + (l*l + ml) * FDIM + n] = c[r];
            }
        }
    }
}

// ---- kernel 2: radial basis + W3 + sparse real-CG tensor product -----------
#define TPB2 64
__global__ __launch_bounds__(TPB2) void k2_tp(
        const float* __restrict__ disp, const float* __restrict__ W3,
        const float* __restrict__ b3,   const float* __restrict__ wtp,
        const Ent* __restrict__ entries, const int* __restrict__ nnzPtr,
        float* __restrict__ out) {
    __shared__ float rad[FDIM];
    __shared__ float sBD[9  * TPB2];
    __shared__ float sY [25 * TPB2];
    __shared__ float sO [25 * TPB2];

    const int e = blockIdx.x >> 1;
    const int g = ((blockIdx.x & 1) << 6) | threadIdx.x;
    const int t = threadIdx.x;

    float dx = disp[e*3+0], dy = disp[e*3+1], dz = disp[e*3+2];
    float r  = sqrtf(dx*dx + dy*dy + dz*dz + 1e-12f);
    float uc = r / CUTOFF_R;
    float denom = fmaxf(1.f - uc*uc, 1e-9f);
    float cut = (uc < 1.f) ? __expf(1.f - 1.f/denom) : 0.f;
    const float pioc = 3.14159265358979323846f / CUTOFF_R;
    for (int k = t; k < FDIM; k += TPB2)
        rad[k] = cosf((float)k * (pioc * r)) * cut;
    __syncthreads();

    float rw0 = 0.f, rw1 = 0.f, rw2 = 0.f;
    for (int k = 0; k < FDIM; ++k) {
        float rk = rad[k];
        rw0 += rk * W3[(0*FDIM + k) * FDIM + g];
        rw1 += rk * W3[(1*FDIM + k) * FDIM + g];
        rw2 += rk * W3[(2*FDIM + k) * FDIM + g];
    }

    float invr = 1.f / r;
    float x = dx*invr, y = dy*invr, z = dz*invr;
    const float s3 = 1.7320508075688772f;
    float sh[9] = {1.f, y, z, x, s3*x*y, s3*y*z, 0.5f*(3.f*z*z - 1.f),
                   s3*x*z, 0.5f*s3*(x*x - y*y)};
    sBD[0*TPB2 + t] = sh[0]*rw0 + b3[g];
#pragma unroll
    for (int a = 1; a < 4; ++a) sBD[a*TPB2 + t] = sh[a]*rw1;
#pragma unroll
    for (int a = 4; a < 9; ++a) sBD[a*TPB2 + t] = sh[a]*rw2;
#pragma unroll
    for (int b = 0; b < 25; ++b) sY[b*TPB2 + t] = out[(size_t)e*3200 + b*FDIM + g];
#pragma unroll
    for (int c = 0; c < 25; ++c) sO[c*TPB2 + t] = 0.f;

    int nnz = *nnzPtr;
    for (int q = 0; q < nnz; ++q) {
        float v = entries[q].val;
        int pk  = entries[q].packed;
        int a = pk & 255, b = (pk >> 8) & 255, c = (pk >> 16) & 255, p = (pk >> 24) & 255;
        sO[c*TPB2 + t] += v * wtp[p*FDIM + g] * sBD[a*TPB2 + t] * sY[b*TPB2 + t];
    }
#pragma unroll
    for (int c = 0; c < 25; ++c)
        out[(size_t)e*3200 + c*FDIM + g] = sO[c*TPB2 + t];
}

// ---- launch ----------------------------------------------------------------
extern "C" void kernel_launch(void* const* d_in, const int* in_sizes, int n_in,
                              void* d_out, int out_size, void* d_ws, size_t ws_size,
                              hipStream_t stream) {
    const float* Adesc = (const float*)d_in[0];
    const int*   nbr   = (const int*)  d_in[1];
    const float* disp  = (const float*)d_in[2];
    const float* W1    = (const float*)d_in[3];
    const float* b1    = (const float*)d_in[4];
    const float* W2    = (const float*)d_in[5];
    const float* b2    = (const float*)d_in[6];
    const float* W3    = (const float*)d_in[7];
    const float* b3    = (const float*)d_in[8];
    const float* wtp   = (const float*)d_in[9];
    float* out = (float*)d_out;

    int nEdges = in_sizes[1] / 2;             // 16384

    float* cgDense = (float*)d_ws;                         // 2391 floats used
    int*   nnzCnt  = (int*)((char*)d_ws + 10240);
    Ent*   entries = (Ent*)((char*)d_ws + 10256);

    k0_cg<<<NPATH, 512, 0, stream>>>(cgDense);
    k0_compact<<<1, 1, 0, stream>>>(cgDense, entries, nnzCnt);

    (void)hipFuncSetAttribute(reinterpret_cast<const void*>(k1_edense),
                              hipFuncAttributeMaxDynamicSharedMemorySize, K1_LDS);
    k1_edense<<<nEdges / 8, 256, K1_LDS, stream>>>(Adesc, nbr, W1, b1, W2, b2, out);
    k2_tp<<<nEdges * 2, TPB2, 0, stream>>>(disp, W3, b3, wtp, entries, nnzCnt, out);
}
